// BayesianGNN_18717467476491
// MI455X (gfx1250) — matrix operand order repaired and checked by reading, not compiled, not run on hardware
//
#include <hip/hip_runtime.h>

// ---------------------------------------------------------------------------
// BayesianGNN forward for MI455X (gfx1250, wave32, WMMA).
//
// Input flattening assumption (setup_inputs dict insertion order; params as a
// JAX pytree with sorted dict keys, lists in order):
//   0:nodes 1:edges 2:globals_
//   3..4   edge_embed   {b,w}
//   5..28  edge_mlp     2 steps x 3 layers x {bm,br,wm,wr}
//   29..30 global_embed {b,w}
//   31..54 global_mlp
//   55..56 node_embed   {b,w}
//   57..80 node_mlp
//   81..92 readout      3 layers x {bm,br,wm,wr}
//   93:senders 94:receivers   (unused by the reference math)
//
// Output: [ readout(16) | n(100000x128) | e(400000x128) ] as f32.
//
// Workspace layout (bytes):   needs ~860 KB
//   0       : packed bf16 WMMA-B weights, 3 entities x 106496 ushorts
//             (embed 64x128 = 8192, then 6 layers x 16384)
//   655360  : biases f32, 3 entities x 7 layers x 128
//   688128  : readout f32: W1[16384] W2[16384] w3[128] b1[128] b2[128] b3[1]
//   851968  : g hidden buffer 16x128 f32
// ---------------------------------------------------------------------------

typedef __attribute__((ext_vector_type(16))) __bf16        v16bf;
typedef __attribute__((ext_vector_type(8)))  float         v8f;
typedef __attribute__((ext_vector_type(8)))  unsigned int  v8u;

#define ENT_W   106496u          // ushorts per entity weight block
#define BIAS_OFF 655360u
#define RO_OFF   688128u
#define GBUF_OFF 851968u

struct PtrTable { const float* p[96]; };

// -------------------- deterministic noise + helpers ------------------------

__device__ __forceinline__ float sp_(float x) {             // softplus
  return (x > 20.0f) ? x : log1pf(__expf(x));
}

__device__ __forceinline__ float nrm_(unsigned int strm, unsigned int idx) {
  // splitmix64 -> Box-Muller; deterministic per (stream, idx), seed = 42
  unsigned long long z = (((unsigned long long)strm << 32) | idx);
  z += 0x9E3779B97F4A7C15ull * 43ull;
  z ^= z >> 30; z *= 0xBF58476D1CE4E5B9ull;
  z ^= z >> 27; z *= 0x94D049BB133111EBull;
  z ^= z >> 31;
  unsigned int u1 = (unsigned int)z;
  unsigned int u2 = (unsigned int)(z >> 32);
  float f1 = ((float)u1 + 1.0f) * 2.3283064365386963e-10f;   // (0,1]
  float f2 = (float)u2 * 2.3283064365386963e-10f;
  return sqrtf(-2.0f * __logf(f1)) * __cosf(6.28318530717958f * f2);
}

__device__ __forceinline__ unsigned short f2bf_(float f) {
  return __builtin_bit_cast(unsigned short, (__bf16)f);
}

// -------------------- weight prep: reparameterize + pack -------------------
// Packed B layout per layer (K x 128):  element (k,n) ->
//   c = k>>5, hh = (k>>3)&1, j = (k&7) | (((k>>4)&1)<<3), t = n>>4, lane = (n&15) | (hh<<4)
//   dst[ ((c*8+t)*32 + lane)*16 + j ]
// so each lane's 16 bf16 B-operand values are 32 contiguous bytes.

__global__ void __launch_bounds__(256)
prep_weights(PtrTable pt, unsigned short* __restrict__ WB,
             float* __restrict__ BIAS, float* __restrict__ RO) {
  const int job = blockIdx.x;
  const int tid = threadIdx.x;

  if (job < 18) {                         // MLP layers, with noise
    const int ent = job / 6;              // 0=edge 1=node 2=global
    const int l   = job % 6;
    const int mb  = (ent == 0) ? 5 : (ent == 1) ? 57 : 31;
    const int li  = mb + (l / 3) * 12 + (l % 3) * 4;
    const float* bm = pt.p[li + 0];
    const float* br = pt.p[li + 1];
    const float* wm = pt.p[li + 2];
    const float* wr = pt.p[li + 3];
    unsigned short* dst = WB + (size_t)ent * ENT_W + 8192u + (size_t)l * 16384u;
    const unsigned int strm = (unsigned int)job;
    for (int e = tid; e < 128 * 128; e += 256) {
      const int k = e >> 7, n = e & 127;
      const float w = wm[e] + nrm_(strm, (unsigned)e) * sp_(wr[e]);
      const int c = k >> 5, kk = k & 31;
      const int hh = (kk >> 3) & 1, j = (kk & 7) | ((kk >> 4) << 3);
      const int t = n >> 4;
      const int lane = (n & 15) | (hh << 4);
      dst[(size_t)((c * 8 + t) * 32 + lane) * 16 + j] = f2bf_(w);
    }
    if (tid < 128) {
      BIAS[ent * 896 + (l + 1) * 128 + tid] =
          bm[tid] + nrm_(strm, 0x40000000u + (unsigned)tid) * sp_(br[tid]);
    }
  } else if (job < 21) {                  // embeds, deterministic
    const int ent = job - 18;
    const int wi = (ent == 0) ? 4 : (ent == 1) ? 56 : 30;
    const float* w = pt.p[wi];
    const float* b = pt.p[wi - 1];
    unsigned short* dst = WB + (size_t)ent * ENT_W;
    for (int e = tid; e < 64 * 128; e += 256) {
      const int k = e >> 7, n = e & 127;
      const int c = k >> 5, kk = k & 31;
      const int hh = (kk >> 3) & 1, j = (kk & 7) | ((kk >> 4) << 3);
      const int t = n >> 4;
      const int lane = (n & 15) | (hh << 4);
      dst[(size_t)((c * 8 + t) * 32 + lane) * 16 + j] = f2bf_(w[e]);
    }
    if (tid < 128) BIAS[ent * 896 + tid] = b[tid];
  } else if (job < 23) {                  // readout L0/L1 -> plain f32
    const int l  = job - 21;
    const int li = 81 + l * 4;
    const float* bm = pt.p[li + 0];
    const float* br = pt.p[li + 1];
    const float* wm = pt.p[li + 2];
    const float* wr = pt.p[li + 3];
    float* W = RO + l * 16384;
    const unsigned int strm = (unsigned int)job;
    for (int e = tid; e < 128 * 128; e += 256)
      W[e] = wm[e] + nrm_(strm, (unsigned)e) * sp_(wr[e]);
    if (tid < 128)
      RO[32896 + l * 128 + tid] =
          bm[tid] + nrm_(strm, 0x40000000u + (unsigned)tid) * sp_(br[tid]);
  } else {                                // readout L2 (128 -> 1)
    const int li = 89;
    const float* bm = pt.p[li + 0];
    const float* br = pt.p[li + 1];
    const float* wm = pt.p[li + 2];
    const float* wr = pt.p[li + 3];
    const unsigned int strm = 23u;
    if (tid < 128)
      RO[32768 + tid] = wm[tid] + nrm_(strm, (unsigned)tid) * sp_(wr[tid]);
    if (tid == 0)
      RO[33152] = bm[0] + nrm_(strm, 0x40000000u) * sp_(br[0]);
  }
}

// -------------------- fused 7-stage WMMA chain -----------------------------

template <int NCH>
__device__ __forceinline__ void mm_stage(const v16bf (&A)[4],
                                         const unsigned short* Bp,
                                         v8f (&acc)[8]) {
#pragma unroll
  for (int c = 0; c < NCH; ++c) {
#pragma unroll
    for (int t = 0; t < 8; ++t) {
      const v8u bw = *(const v8u*)(Bp + (c * 8 + t) * 512);
      const v16bf b = __builtin_bit_cast(v16bf, bw);
      acc[t] = __builtin_amdgcn_wmma_f32_16x16x32_bf16(
          false, A[c], false, b, (short)0, acc[t], false, false);
    }
  }
}

__device__ __forceinline__ v16bf cvtA8(const float* p) {
  // A-operand chunk half: 8 f32 at +0 (elements 0..7), 8 f32 at +16 (8..15)
  float4 f0 = *(const float4*)(p);
  float4 f1 = *(const float4*)(p + 4);
  float4 g0 = *(const float4*)(p + 16);
  float4 g1 = *(const float4*)(p + 20);
  v16bf a;
  a[0]  = (__bf16)f0.x; a[1]  = (__bf16)f0.y; a[2]  = (__bf16)f0.z; a[3]  = (__bf16)f0.w;
  a[4]  = (__bf16)f1.x; a[5]  = (__bf16)f1.y; a[6]  = (__bf16)f1.z; a[7]  = (__bf16)f1.w;
  a[8]  = (__bf16)g0.x; a[9]  = (__bf16)g0.y; a[10] = (__bf16)g0.z; a[11] = (__bf16)g0.w;
  a[12] = (__bf16)g1.x; a[13] = (__bf16)g1.y; a[14] = (__bf16)g1.z; a[15] = (__bf16)g1.w;
  return a;
}

__global__ void __launch_bounds__(256)
gnn_chain(const float* __restrict__ X, int M, float* __restrict__ Y,
          const unsigned short* __restrict__ WBe, const float* __restrict__ Be) {
  __shared__ float lds[8 * 2048];                 // 64 KB: 8 KB per wave
  const int tid  = threadIdx.x;
  const int wave = tid >> 5;
  const int lane = tid & 31;
  const int half = lane >> 4;
  const int mloc = lane & 15;
  float* L = &lds[wave * 2048];
  const int row0 = (blockIdx.x * 8 + wave) * 16;

  v16bf A[4];
  {                                               // stage-0 A from global f32
    int row = row0 + mloc;
    row = (row < M) ? row : (M - 1);
    const float* xr = X + (size_t)row * 64 + half * 8;
#pragma unroll
    for (int c = 0; c < 2; ++c) A[c] = cvtA8(xr + c * 32);
  }

  for (int s = 0; s < 7; ++s) {                   // embed + 6 MLP layers
    const unsigned short* Bp =
        WBe + ((s == 0) ? 0u : (8192u + (unsigned)(s - 1) * 16384u)) + lane * 16;
    if (s < 6)                                    // warm next stage's weights
      __builtin_prefetch(WBe + 8192u + (unsigned)s * 16384u + lane * 16, 0, 1);

    v8f acc[8];
#pragma unroll
    for (int t = 0; t < 8; ++t) { v8f z = {}; acc[t] = z; }

    if (s == 0) mm_stage<2>(A, Bp, acc);
    else        mm_stage<4>(A, Bp, acc);

    const bool relu = (s == 1) | (s == 2) | (s == 4) | (s == 5);
#pragma unroll
    for (int t = 0; t < 8; ++t) {
      const float bv = Be[s * 128 + t * 16 + mloc];
#pragma unroll
      for (int r = 0; r < 8; ++r) {
        float v = acc[t][r] + bv;
        acc[t][r] = relu ? fmaxf(v, 0.0f) : v;
      }
    }

    // D layout (lane = N column, VGPR = M row) -> row-major f32 tile in LDS
#pragma unroll
    for (int t = 0; t < 8; ++t)
#pragma unroll
      for (int r = 0; r < 8; ++r)
        L[(r + half * 8) * 128 + t * 16 + mloc] = acc[t][r];
    __syncthreads();

    if (s < 6) {                                  // reload as next A operands
#pragma unroll
      for (int c = 0; c < 4; ++c)
        A[c] = cvtA8(L + mloc * 128 + c * 32 + half * 8);
    }
  }

  // coalesced f32 output of a 64-row x 32-col block read from L (row stride 128)
#pragma unroll
  for (int i = 0; i < 16; ++i) {
    const int idx = i * 32 + lane;
    const int row = idx >> 3;
    const int c4  = (idx & 7) * 4;
    if (row0 + row < M)
      *(float4*)(Y + (size_t)(row0 + row) * 128 + c4) =
          *(const float4*)(L + row * 128 + c4);
  }
}

// -------------------- tiny scalar readout (16 rows) ------------------------

__global__ void __launch_bounds__(128)
readout_kernel(const float* __restrict__ G, const float* __restrict__ RO,
               float* __restrict__ out) {
  __shared__ float a[16 * 128];
  __shared__ float h[16 * 128];
  const int tid = threadIdx.x;
  for (int i = tid; i < 2048; i += 128) a[i] = G[i];
  __syncthreads();
  const float* W1 = RO;
  const float* W2 = RO + 16384;
  const float* w3 = RO + 32768;
  const float* b1 = RO + 32896;
  const float* b2 = RO + 33024;
  const float* b3 = RO + 33152;
  for (int m = 0; m < 16; ++m) {
    float acc = b1[tid];
    for (int k = 0; k < 128; ++k) acc += a[m * 128 + k] * W1[k * 128 + tid];
    h[m * 128 + tid] = fmaxf(acc, 0.0f);
  }
  __syncthreads();
  for (int m = 0; m < 16; ++m) {
    float acc = b2[tid];
    for (int k = 0; k < 128; ++k) acc += h[m * 128 + k] * W2[k * 128 + tid];
    a[m * 128 + tid] = fmaxf(acc, 0.0f);
  }
  __syncthreads();
  if (tid < 16) {
    float acc = b3[0];
    for (int k = 0; k < 128; ++k) acc += a[tid * 128 + k] * w3[k];
    out[tid] = acc;
  }
}

// -------------------- launch -----------------------------------------------

extern "C" void kernel_launch(void* const* d_in, const int* in_sizes, int n_in,
                              void* d_out, int out_size, void* d_ws, size_t ws_size,
                              hipStream_t stream) {
  (void)in_sizes; (void)out_size; (void)ws_size;

  PtrTable pt;
  for (int i = 0; i < 96; ++i)
    pt.p[i] = (i < n_in) ? (const float*)d_in[i] : nullptr;

  const float* nodes    = (const float*)d_in[0];
  const float* edges    = (const float*)d_in[1];
  const float* globals_ = (const float*)d_in[2];

  unsigned short* WB = (unsigned short*)d_ws;
  float* BIAS = (float*)((char*)d_ws + BIAS_OFF);
  float* RO   = (float*)((char*)d_ws + RO_OFF);
  float* GBUF = (float*)((char*)d_ws + GBUF_OFF);

  float* out   = (float*)d_out;
  float* n_out = out + 16;
  float* e_out = out + 16 + (size_t)100000 * 128;

  prep_weights<<<24, 256, 0, stream>>>(pt, WB, BIAS, RO);

  gnn_chain<<<(400000 + 127) / 128, 256, 0, stream>>>(
      edges, 400000, e_out, WB + (size_t)0 * ENT_W, BIAS + 0 * 896);
  gnn_chain<<<(100000 + 127) / 128, 256, 0, stream>>>(
      nodes, 100000, n_out, WB + (size_t)1 * ENT_W, BIAS + 1 * 896);
  gnn_chain<<<1, 256, 0, stream>>>(
      globals_, 16, GBUF, WB + (size_t)2 * ENT_W, BIAS + 2 * 896);

  readout_kernel<<<1, 128, 0, stream>>>(GBUF, RO, out);
}
